// DGMoE_15315853378160
// MI455X (gfx1250) — compile-verified, matrix operand order checked by reading, not run on hardware
//
#include <hip/hip_runtime.h>

typedef __attribute__((ext_vector_type(16))) __bf16 v16bf;
typedef __attribute__((ext_vector_type(8)))  __bf16 v8bf;
typedef __attribute__((ext_vector_type(4)))  __bf16 v4bf;
typedef __attribute__((ext_vector_type(8)))  float  v8f;

#define DD 1024
#define FF 4096
#define KK 8
#define NTOK 4096          // B*T
#define MT 32              // token tile rows per block (2 x 16 subtiles)
#define FC 128             // f-chunk
#define XROW 1032          // padded LDS row (bf16): 2064B -> 4-bank shift/row
#define HROW 136           // padded Hc row (bf16):  272B -> 4-bank shift/row

__device__ __forceinline__ float gelu_exact(float v) {
    return 0.5f * v * (1.0f + erff(v * 0.70710678118654752f));
}

__device__ __forceinline__ v16bf cat16(v8bf lo, v8bf hi) {
    return __builtin_shufflevector(lo, hi, 0,1,2,3,4,5,6,7,8,9,10,11,12,13,14,15);
}

// ---------------------------------------------------------------------------
// Transpose last-two-dims + fp32 -> bf16 convert:  in[k][r][c] -> out[k][c][r]
// ---------------------------------------------------------------------------
__global__ __launch_bounds__(256) void transpose_bf16_kernel(
    const float* __restrict__ in, __bf16* __restrict__ out, int R, int C)
{
    __shared__ float tile[32][33];
    const int k  = blockIdx.z;
    const int c0 = blockIdx.x * 32;
    const int r0 = blockIdx.y * 32;
    const float* inp = in + (size_t)k * R * C;
    __bf16*     outp = out + (size_t)k * R * C;
    const int tx = threadIdx.x, ty = threadIdx.y;
    #pragma unroll
    for (int i = 0; i < 32; i += 8)
        tile[ty + i][tx] = inp[(size_t)(r0 + ty + i) * C + (c0 + tx)];
    __syncthreads();
    #pragma unroll
    for (int i = 0; i < 32; i += 8)
        outp[(size_t)(c0 + ty + i) * R + (r0 + tx)] = (__bf16)tile[tx][ty + i];
}

// ---------------------------------------------------------------------------
// Router + gating: logits, softmax over K=8, threshold mask, g, 1/(sum g+eps),
// sel_counts and avg_density. 8 lanes per token, 32 tokens per block.
// ---------------------------------------------------------------------------
__global__ __launch_bounds__(256) void router_kernel(
    const float* __restrict__ x, const float* __restrict__ prev,
    const float* __restrict__ Wt, const float* __restrict__ Wgt,
    const float* __restrict__ We,
    float* __restrict__ logits_out, float* __restrict__ gbuf,
    float* __restrict__ ginv, float* __restrict__ sel, float* __restrict__ dens)
{
    const int tid   = threadIdx.x;
    const int token = blockIdx.x * 32 + (tid >> 3);
    const int k     = tid & 7;

    const float* xr = x + (size_t)token * DD;
    float acc = 0.0f;
    #pragma unroll 4
    for (int d = 0; d < DD; ++d)
        acc += xr[d] * Wt[d * KK + k];
    #pragma unroll
    for (int j = 0; j < KK; ++j)
        acc += prev[(size_t)token * KK + j] * Wgt[j * KK + k];

    logits_out[(size_t)token * KK + k] = acc;

    float m = acc;
    m = fmaxf(m, __shfl_xor(m, 1, 8));
    m = fmaxf(m, __shfl_xor(m, 2, 8));
    m = fmaxf(m, __shfl_xor(m, 4, 8));
    float e = expf(acc - m);
    float s = e;
    s += __shfl_xor(s, 1, 8);
    s += __shfl_xor(s, 2, 8);
    s += __shfl_xor(s, 4, 8);
    const float st  = e / s;
    const float thr = 0.5f / (1.0f + expf(-We[k]));   // LAMBDA * sigmoid(We)
    const float mk  = (st > thr) ? 1.0f : 0.0f;
    const float g   = st * mk;

    float gsum = g;
    gsum += __shfl_xor(gsum, 1, 8);
    gsum += __shfl_xor(gsum, 2, 8);
    gsum += __shfl_xor(gsum, 4, 8);
    float msum = mk;
    msum += __shfl_xor(msum, 1, 8);
    msum += __shfl_xor(msum, 2, 8);
    msum += __shfl_xor(msum, 4, 8);

    gbuf[(size_t)token * KK + k] = g;
    if (k == 0) ginv[token] = 1.0f / (gsum + 1e-6f);
    atomicAdd(&sel[k], mk);
    if (k == 0) atomicAdd(dens, msum * (1.0f / (float)NTOK));
}

// ---------------------------------------------------------------------------
// Fused MoE FFN, 32-token tile. Wave w: mt = w&1 (16-row subtile),
// nq = w>>1 (256-col quarter of D; also selects its pair of Hc f-subtiles).
//   H = gelu(X @ W1 + b1) * g[:,k]   (gate pre-scaled into H)
//   Y += H @ W2  over all f-chunks and experts (single accumulator)
// Epilogue: y = (Y + sum_k g_k*b2_k) * ginv.  bf16 WMMA, f32 accumulate.
// ---------------------------------------------------------------------------
__global__ __launch_bounds__(256) void moe_ffn_kernel(
    const float* __restrict__ x,
    const __bf16* __restrict__ w1t,   // (K,F,D) bf16  (transposed W1)
    const float* __restrict__ b1g,    // (K,F)
    const __bf16* __restrict__ w2t,   // (K,D,F) bf16  (transposed W2)
    const float* __restrict__ b2g,    // (K,D)
    const float* __restrict__ gbuf,   // (NTOK,K)
    const float* __restrict__ ginv,   // (NTOK)
    float* __restrict__ yout)         // (NTOK,D)
{
    __shared__ __attribute__((aligned(16))) __bf16 Xs[MT][XROW];
    __shared__ __attribute__((aligned(16))) __bf16 Hcs[MT][HROW];
    __shared__ float gs[MT][KK];
    __shared__ float gis[MT];
    __shared__ int   eflag[KK];

    const int tid  = threadIdx.x;
    const int wave = tid >> 5;
    const int lane = tid & 31;
    const int ln16 = lane & 15;
    const int hoff = (lane >> 4) << 3;     // 0 for lanes 0-15, 8 for 16-31
    const int mt   = (wave & 1) << 4;      // row subtile base (0 or 16)
    const int nq   = wave >> 1;            // D-quarter (0..3)
    const int tok0 = blockIdx.x * MT;

    // stage X tile as bf16 in LDS (vectorized: float4 in, 8B bf16x4 out)
    {
        const float4* xv = (const float4*)x;
        for (int i = tid; i < (MT * DD) / 4; i += 256) {
            const int r = i >> 8;             // DD/4 = 256 quads per row
            const int c = (i & 255) << 2;
            float4 vv = xv[((size_t)(tok0 + r) * DD + c) >> 2];
            v4bf o = { (__bf16)vv.x, (__bf16)vv.y, (__bf16)vv.z, (__bf16)vv.w };
            *(v4bf*)&Xs[r][c] = o;
        }
    }
    if (tid < MT * KK) {
        const int r = tid >> 3, k = tid & 7;
        gs[r][k] = gbuf[(size_t)(tok0 + r) * KK + k];
    }
    if (tid < MT) gis[tid] = ginv[tok0 + tid];
    __syncthreads();
    if (tid < KK) {
        int any = 0;
        for (int r = 0; r < MT; ++r) any |= (gs[r][tid] != 0.0f) ? 1 : 0;
        eflag[tid] = any;
    }
    __syncthreads();

    v8f yacc[16];
    #pragma unroll
    for (int t = 0; t < 16; ++t) yacc[t] = v8f{0.f,0.f,0.f,0.f,0.f,0.f,0.f,0.f};

    for (int k = 0; k < KK; ++k) {
        if (!eflag[k]) continue;           // block-uniform skip
        for (int fc = 0; fc < FF; fc += FC) {
            // ---- GEMM1: wave computes Hc rows [mt,mt+16), subtiles 2nq,2nq+1
            for (int s2 = 0; s2 < 2; ++s2) {
                const int sft  = (nq << 1) + s2;            // f-subtile 0..7
                const int fcol = fc + (sft << 4) + ln16;    // B-lane column
                const __bf16* w1p = w1t + ((size_t)k * FF + fcol) * DD;
                v8f hacc = v8f{0.f,0.f,0.f,0.f,0.f,0.f,0.f,0.f};
                #pragma unroll 4
                for (int d = 0; d < DD; d += 32) {
                    const __bf16* xp = &Xs[mt + ln16][d + hoff];
                    v16bf A = cat16(*(const v8bf*)xp, *(const v8bf*)(xp + 16));
                    const __bf16* bp = w1p + d + hoff;
                    v16bf Bm = cat16(*(const v8bf*)bp, *(const v8bf*)(bp + 16));
                    hacc = __builtin_amdgcn_wmma_f32_16x16x32_bf16(
                        false, A, false, Bm, (short)0, hacc, false, false);
                }
                const float bias = b1g[(size_t)k * FF + fcol];
                if (s2 == 0) __syncthreads();   // WAR: prior GEMM2 reads done
                #pragma unroll
                for (int r = 0; r < 8; ++r) {
                    const int row = mt + r + hoff;   // C layout row mapping
                    const float h = gelu_exact(hacc[r] + bias) * gs[row][k];
                    Hcs[row][(sft << 4) + ln16] = (__bf16)h;
                }
            }
            __syncthreads();
            // ---- GEMM2: Y(16 x 256 per wave) += Hc(16xFC) @ W2t ----
            #pragma unroll
            for (int ks = 0; ks < FC; ks += 32) {
                const __bf16* hp = &Hcs[mt + ln16][ks + hoff];
                v16bf A = cat16(*(const v8bf*)hp, *(const v8bf*)(hp + 16));
                #pragma unroll
                for (int t = 0; t < 16; ++t) {
                    const int dcol = (nq << 8) + (t << 4) + ln16;
                    const __bf16* w2p =
                        w2t + ((size_t)k * DD + dcol) * FF + fc + ks + hoff;
                    v16bf Bm = cat16(*(const v8bf*)w2p, *(const v8bf*)(w2p + 16));
                    yacc[t] = __builtin_amdgcn_wmma_f32_16x16x32_bf16(
                        false, A, false, Bm, (short)0, yacc[t], false, false);
                }
            }
        }
    }

    // ---- epilogue: add sum_k g_k*b2_k, scale by 1/(sum g + eps), store ----
    #pragma unroll
    for (int t = 0; t < 16; ++t) {
        const int dcol = (nq << 8) + (t << 4) + ln16;
        float b2v[KK];
        #pragma unroll
        for (int kk = 0; kk < KK; ++kk) b2v[kk] = b2g[(size_t)kk * DD + dcol];
        #pragma unroll
        for (int r = 0; r < 8; ++r) {
            const int row = mt + r + hoff;
            float bias2 = 0.0f;
            #pragma unroll
            for (int kk = 0; kk < KK; ++kk) bias2 += gs[row][kk] * b2v[kk];
            yout[(size_t)(tok0 + row) * DD + dcol] = (yacc[t][r] + bias2) * gis[row];
        }
    }
}

// ---------------------------------------------------------------------------
extern "C" void kernel_launch(void* const* d_in, const int* in_sizes, int n_in,
                              void* d_out, int out_size, void* d_ws, size_t ws_size,
                              hipStream_t stream) {
    const float* x    = (const float*)d_in[0];
    const float* prev = (const float*)d_in[1];
    const float* Wt   = (const float*)d_in[2];
    const float* Wgt  = (const float*)d_in[3];
    const float* We   = (const float*)d_in[4];
    const float* W1   = (const float*)d_in[5];
    const float* b1   = (const float*)d_in[6];
    const float* W2   = (const float*)d_in[7];
    const float* b2   = (const float*)d_in[8];
    float* out = (float*)d_out;

    const size_t wbytes = (size_t)KK * DD * FF * sizeof(__bf16);   // 64 MB each
    const size_t need   = 2 * wbytes + (size_t)NTOK * (KK + 1) * sizeof(float);
    if (ws_size < need) return;                                    // ws too small

    __bf16* w1t = (__bf16*)d_ws;
    __bf16* w2t = (__bf16*)((char*)d_ws + wbytes);
    float*  gbuf = (float*)((char*)d_ws + 2 * wbytes);
    float*  gin  = gbuf + (size_t)NTOK * KK;

    const size_t Yoff = (size_t)NTOK * DD;            // y: 4,194,304 floats
    const size_t Loff = Yoff;                         // logits: 32,768 floats
    const size_t Soff = Yoff + (size_t)NTOK * KK;     // sel_counts: 8 floats
    const size_t Aoff = Soff + KK;                    // avg_density: 1 float

    hipMemsetAsync(out + Soff, 0, (KK + 1) * sizeof(float), stream);

    // one-time weight convert+transpose into L2-resident bf16 (134 MB < 192 MB L2)
    transpose_bf16_kernel<<<dim3(FF / 32, DD / 32, KK), dim3(32, 8), 0, stream>>>(
        W1, w1t, DD, FF);
    transpose_bf16_kernel<<<dim3(DD / 32, FF / 32, KK), dim3(32, 8), 0, stream>>>(
        W2, w2t, FF, DD);

    router_kernel<<<NTOK / 32, 256, 0, stream>>>(
        x, prev, Wt, Wgt, We, out + Loff, gbuf, gin, out + Soff, out + Aoff);

    moe_ffn_kernel<<<NTOK / MT, 256, 0, stream>>>(
        x, w1t, b1, w2t, b2, gbuf, gin, out);
}
